// CondConv_37469294690898
// MI455X (gfx1250) — matrix operand order repaired and checked by reading, not compiled, not run on hardware
//
#include <hip/hip_runtime.h>
#include <math.h>

typedef __attribute__((ext_vector_type(16))) _Float16 v16h;
typedef __attribute__((ext_vector_type(8)))  _Float16 v8h;
typedef __attribute__((ext_vector_type(8)))  float    v8f;

#define B_    32
#define CIN_  256
#define H_    56
#define W_    56
#define HW_   3136
#define COUT_ 256
#define E_    8

// ---------------- routing: GAP -> linear -> sigmoid ----------------

__global__ __launch_bounds__(256) void gap_kernel(const float* __restrict__ x,
                                                  float* __restrict__ gap) {
  __shared__ float red[256];
  int bc = blockIdx.x;                      // b*CIN + c
  const float* p = x + (size_t)bc * HW_;
  float s = 0.f;
  for (int i = threadIdx.x; i < HW_; i += 256) s += p[i];
  red[threadIdx.x] = s;
  __syncthreads();
  for (int off = 128; off > 0; off >>= 1) {
    if (threadIdx.x < off) red[threadIdx.x] += red[threadIdx.x + off];
    __syncthreads();
  }
  if (threadIdx.x == 0) gap[bc] = red[0] * (1.0f / (float)HW_);
}

__global__ __launch_bounds__(256) void routing_kernel(const float* __restrict__ gap,
                                                      const float* __restrict__ Wa,
                                                      const float* __restrict__ ba,
                                                      float* __restrict__ routing) {
  int t = threadIdx.x;                      // 256 = 32 b * 8 e
  int b = t >> 3, e = t & 7;
  float z = ba[e];
  const float* g = gap + b * CIN_;
  const float* w = Wa + e * CIN_;
  for (int c = 0; c < CIN_; ++c) z += g[c] * w[c];
  routing[t] = 1.0f / (1.0f + expf(-z));
}

// -------- x transpose+convert: x[b][c][h][w] f32 -> xt[b][h][w][c] f16 --------

__global__ __launch_bounds__(256) void xt_kernel(const float* __restrict__ x,
                                                 _Float16* __restrict__ xt) {
  __shared__ _Float16 tile[64][57];
  int b = blockIdx.y, h = blockIdx.x;
  for (int c0 = 0; c0 < CIN_; c0 += 64) {
    for (int i = threadIdx.x; i < 64 * W_; i += 256) {
      int cl = i / W_, w = i - cl * W_;
      tile[cl][w] = (_Float16)x[(((size_t)b * CIN_ + c0 + cl) * H_ + h) * W_ + w];
    }
    __syncthreads();
    for (int i = threadIdx.x; i < W_ * 64; i += 256) {
      int w = i >> 6, cl = i & 63;
      xt[((size_t)b * HW_ + h * W_ + w) * CIN_ + c0 + cl] = tile[cl][w];
    }
    __syncthreads();
  }
}

// -------- expert mixing into WMMA-A layout: kA[b][p][cc][cout][kk] f16 --------
// p = ky*3+kx (0..8), cc = cin chunk of 32 (0..7), kk = cin within chunk

__global__ __launch_bounds__(256) void mix_kernel(const float* __restrict__ convs,
                                                  const float* __restrict__ routing,
                                                  _Float16* __restrict__ kA) {
  int cc = blockIdx.x;        // 0..7
  int p  = blockIdx.y;        // 0..8
  int b  = blockIdx.z;        // 0..31
  float r[E_];
  for (int e = 0; e < E_; ++e) r[e] = routing[b * E_ + e];
  _Float16* outp = kA + ((((size_t)b * 9 + p) * 8 + cc) << 13);   // *256*32
  for (int j = threadIdx.x; j < 8192; j += 256) {
    int cout = j >> 5, kk = j & 31;
    int cin = (cc << 5) + kk;
    int base = (cout * CIN_ + cin) * 9 + p;
    float s = 0.f;
    #pragma unroll
    for (int e = 0; e < E_; ++e)
      s += r[e] * convs[(size_t)e * (COUT_ * CIN_ * 9) + base];
    outp[j] = (_Float16)s;
  }
}

// ---------------- implicit-GEMM conv via v_wmma_f32_16x16x32_f16 ----------------
// D[cout=256, pos=3136] = kA[cout, 2304] x im2col(xt)[2304, pos]  per sample.
// WG: 8 waves, 4(M)x2(N); each wave: 2(M)x4(N) subtiles of 16x16 -> 32x64.
// WG tile 128 cout x 128 positions. K loop: 9 taps x 8 cin-chunks of 32.
// 12 b128 loads per 8 WMMAs per K-chunk (~21 FLOP/byte from cache).

__global__ __launch_bounds__(256) void condconv_wmma(const _Float16* __restrict__ xt,
                                                     const _Float16* __restrict__ kA,
                                                     float* __restrict__ out) {
  int b  = blockIdx.z;
  int m0 = blockIdx.x * 128;          // cout tile
  int n0 = blockIdx.y * 128;          // position tile (flat h*56+w)
  int wave = threadIdx.x >> 5;
  int lane = threadIdx.x & 31;
  int mBase = m0 + (wave >> 1) * 32;            // 4 M-rows of waves
  int nBase = n0 + (wave & 1) * 64;             // 2 N-cols of waves
  int lmod = lane & 15;
  int hi   = lane >> 4;               // 0|1
  int kA_half = hi * 16;              // A operand: 16 contiguous K at 0|16
  int kB_half = hi * 16;              // B operand: 16 contiguous K at 0|16

  v8f acc[2][4] = {};

  const _Float16* kAb = kA + (size_t)b * (9 * 8 * 256 * 32);
  const _Float16* xtb = xt + (size_t)b * HW_ * CIN_;

  for (int p = 0; p < 9; ++p) {
    int ky = p / 3 - 1, kx = p % 3 - 1;
    int boff[4];
    bool bval[4];
    #pragma unroll
    for (int t = 0; t < 4; ++t) {
      int n = nBase + t * 16 + lmod;
      int h = n / 56, w = n - h * 56;
      int hh = h + ky, ww = w + kx;
      bval[t] = (n < HW_) & (hh >= 0) & (hh < 56) & (ww >= 0) & (ww < 56);
      boff[t] = (hh * 56 + ww) * CIN_ + kB_half;
    }
    const _Float16* Ap = kAb + (size_t)p * (8 * 256 * 32);
    #pragma unroll
    for (int cc = 0; cc < 8; ++cc) {
      const _Float16* Acc = Ap + (cc << 13);
      v16h a[2];
      #pragma unroll
      for (int s = 0; s < 2; ++s) {
        const _Float16* ap = Acc + (mBase + s * 16 + lmod) * 32 + kA_half;
        ((v8h*)&a[s])[0] = *(const v8h*)ap;          // K = kA_half .. +7
        ((v8h*)&a[s])[1] = *(const v8h*)(ap + 8);    // K = kA_half+8 .. +15
      }
      v16h bv[4];
      #pragma unroll
      for (int t = 0; t < 4; ++t) {
        v16h tmp = {};
        if (bval[t]) tmp = *(const v16h*)(xtb + boff[t] + (cc << 5));
        bv[t] = tmp;
      }
      #pragma unroll
      for (int s = 0; s < 2; ++s)
        #pragma unroll
        for (int t = 0; t < 4; ++t)
          acc[s][t] = __builtin_amdgcn_wmma_f32_16x16x32_f16(
              false, a[s], false, bv[t], (short)0, acc[s][t], false, false);
    }
  }

  float* outb = out + (size_t)b * COUT_ * HW_;
  #pragma unroll
  for (int s = 0; s < 2; ++s) {
    #pragma unroll
    for (int t = 0; t < 4; ++t) {
      int n = nBase + t * 16 + lmod;
      if (n < HW_) {
        #pragma unroll
        for (int r = 0; r < 8; ++r) {
          int m = mBase + s * 16 + hi * 8 + r;   // C/D layout: VGPR r, lanes>=16 -> M+8
          outb[(size_t)m * HW_ + n] = acc[s][t][r];
        }
      }
    }
  }
}

// ---------------- launcher ----------------

extern "C" void kernel_launch(void* const* d_in, const int* in_sizes, int n_in,
                              void* d_out, int out_size, void* d_ws, size_t ws_size,
                              hipStream_t stream) {
  const float* x     = (const float*)d_in[0];   // [32,256,56,56]
  const float* convs = (const float*)d_in[1];   // [8,256,256,3,3]
  const float* Wa    = (const float*)d_in[2];   // [8,256]
  const float* ba    = (const float*)d_in[3];   // [8]
  float* out = (float*)d_out;                   // [32,256,56,56] f32

  char* ws = (char*)d_ws;
  float*    gap     = (float*)ws;                               // 32 KB
  float*    routing = (float*)(ws + 32768);                     // 1 KB
  _Float16* xt      = (_Float16*)(ws + 65536);                  // 51,380,224 B
  _Float16* kA      = (_Float16*)(ws + 65536 + 51380224);       // 37,748,736 B

  gap_kernel<<<B_ * CIN_, 256, 0, stream>>>(x, gap);
  routing_kernel<<<1, 256, 0, stream>>>(gap, Wa, ba, routing);
  xt_kernel<<<dim3(H_, B_), 256, 0, stream>>>(x, xt);
  mix_kernel<<<dim3(8, 9, B_), 256, 0, stream>>>(convs, routing, kA);
  condconv_wmma<<<dim3(COUT_ / 128, (HW_ + 127) / 128, B_), 256, 0, stream>>>(xt, kA, out);
}
